// MaskedSelfAttention_18322330484957
// MI455X (gfx1250) — matrix operand order repaired and checked by reading, not compiled, not run on hardware
//
#include <hip/hip_runtime.h>
#include <hip/hip_bf16.h>

// ---------------------------------------------------------------------------
// Masked self-attention for MI455X (gfx1250, wave32, WMMA 16x16x32 bf16)
//   x:[8,2048,512] f32 ; wq/wk/wv:[512,512] f32 ; bq/bk/bv:[512] f32
//   out = softmax(mask(q k^T / sqrt(d))) v      -> d_out[0 .. B*S*D)
//   attn                                        -> d_out[B*S*D .. +B*S*S)
//
// All GEMM fragment loads are contiguous (b128-class) bf16:
//   - weights pre-transposed to bf16 wT[n][k]   (one-time 1.5 MB kernel)
//   - V produced transposed  vT[b][n][s]        (scatter paid once)
//   - softmax emits a bf16 copy of attn so the attn@V GEMM reads bf16
// Register blocking: 8 WMMAs per K-step per wave in every GEMM.
// ---------------------------------------------------------------------------

typedef __attribute__((ext_vector_type(16))) __bf16 v16bf;
typedef __attribute__((ext_vector_type(8)))  float  v8f;

#define BATCH 8
#define SEQ   2048
#define EMBED 512
#define QK_ELEMS ((size_t)BATCH * SEQ * EMBED)   // elements per q/k/v matrix
#define W_ELEMS  ((size_t)EMBED * EMBED)         // elements per weight matrix

#define WMMA_BF16(a, b, c) \
    __builtin_amdgcn_wmma_f32_16x16x32_bf16(false, (a), false, (b), (short)0, (c), false, false)

// ---------------------------------------------------------------------------
// Kernel 0: wT[which][n][k] = (bf16) w[k][n]   (one-time, 3 x 512 x 512)
// grid (512*512/256=1024, 1, 3), block 256
// ---------------------------------------------------------------------------
__global__ __launch_bounds__(256)
void wt_kernel(const float* __restrict__ wq, const float* __restrict__ wk,
               const float* __restrict__ wv, __bf16* __restrict__ wt) {
    const int which = blockIdx.z;
    const float* w = (which == 0) ? wq : (which == 1) ? wk : wv;
    const size_t idx = (size_t)blockIdx.x * 256 + threadIdx.x;  // n*512 + k
    const int n = (int)(idx >> 9);
    const int k = (int)(idx & 511);
    wt[(size_t)which * W_ELEMS + idx] = (__bf16)w[(size_t)k * EMBED + n];
}

// ---------------------------------------------------------------------------
// Kernel 1: q/k/v = x @ W + b  (scale 1/sqrt(d) folded into q), bf16 outputs.
// q,k stored row-major; v stored TRANSPOSED per batch: vT[b][n][s].
// Wave tile: 64(M) x 32(N) -> 4 A + 2 B frags feed 8 WMMAs per K-step.
// grid (EMBED/128=4, M/128=128, 3), block 256 (waves: 2 Msub x 4 Nsub)
// ---------------------------------------------------------------------------
__global__ __launch_bounds__(256)
void qkv_proj_kernel(const float* __restrict__ x,
                     const __bf16* __restrict__ wt,
                     const float* __restrict__ bq,
                     const float* __restrict__ bk,
                     const float* __restrict__ bv,
                     __bf16* __restrict__ qkv_ws) {
    const int lane = threadIdx.x & 31;
    const int wave = threadIdx.x >> 5;
    const int half = lane >> 4;
    const int lo   = lane & 15;

    const int which = blockIdx.z;                 // 0=q 1=k 2=v
    const float* bias = (which == 0) ? bq : (which == 1) ? bk : bv;
    const float scale = (which == 0) ? 0.04419417382415922f : 1.0f;  // 1/sqrt(512)
    const __bf16* wT = wt + (size_t)which * W_ELEMS;   // [N][K] bf16
    __bf16* dst = qkv_ws + (size_t)which * QK_ELEMS;

    const int m0 = blockIdx.y * 128 + (wave & 1) * 64;
    const int n0 = blockIdx.x * 128 + (wave >> 1) * 32;

    v8f c[4][2] = {};
    for (int k0 = 0; k0 < EMBED; k0 += 32) {
        v16bf bfr[2];
#pragma unroll
        for (int nj = 0; nj < 2; ++nj) {
            // B[kk][n] = wT[n0+nj*16+lo][k0+half*8+kk] (kk<8), [k0+16+half*8+kk-8] (kk>=8)
            const __bf16* brow =
                wT + (size_t)(n0 + nj * 16 + lo) * EMBED + k0 + half * 8;
#pragma unroll
            for (int i = 0; i < 8; ++i) { bfr[nj][i] = brow[i]; bfr[nj][i + 8] = brow[16 + i]; }
        }
#pragma unroll
        for (int mi = 0; mi < 4; ++mi) {
            v16bf a;
            const float* arow =
                x + (size_t)(m0 + mi * 16 + lo) * EMBED + k0 + half * 8;
#pragma unroll
            for (int i = 0; i < 8; ++i) {
                a[i]     = (__bf16)arow[i];
                a[i + 8] = (__bf16)arow[16 + i];
            }
#pragma unroll
            for (int nj = 0; nj < 2; ++nj)
                c[mi][nj] = WMMA_BF16(a, bfr[nj], c[mi][nj]);
        }
    }

#pragma unroll
    for (int nj = 0; nj < 2; ++nj) {
        const int col = n0 + nj * 16 + lo;
        const float bn = bias[col];
#pragma unroll
        for (int mi = 0; mi < 4; ++mi)
#pragma unroll
            for (int r = 0; r < 8; ++r) {
                const int row = m0 + mi * 16 + half * 8 + r;  // in [0, B*S)
                const __bf16 val = (__bf16)((c[mi][nj][r] + bn) * scale);
                if (which == 2) {
                    // vT[b][col][s] : transposed store (paid once)
                    const int bb = row >> 11;       // row / SEQ
                    const int s  = row & (SEQ - 1); // row % SEQ
                    dst[((size_t)bb * EMBED + col) * SEQ + s] = val;
                } else {
                    dst[(size_t)row * EMBED + col] = val;
                }
            }
    }
}

// ---------------------------------------------------------------------------
// Kernel 2: raw masked scores -> attn region (fp32).  scores = q_scaled k^T
// Wave tile: 32(M) x 64(N) -> 2 A + 4 B frags feed 8 WMMAs per K-step.
// grid (SEQ/256=8, SEQ/64=32, BATCH), block 256 (waves: 2 Msub x 4 Nsub)
// Fully-masked wave regions are zero-filled (softmax skips -> attn==0).
// ---------------------------------------------------------------------------
__global__ __launch_bounds__(256)
void scores_kernel(const __bf16* __restrict__ qws,
                   const __bf16* __restrict__ kws,
                   float* __restrict__ attn) {
    const int lane = threadIdx.x & 31;
    const int wave = threadIdx.x >> 5;
    const int half = lane >> 4;
    const int lo   = lane & 15;

    const int b  = blockIdx.z;
    const int m0 = blockIdx.y * 64  + (wave & 1) * 32;
    const int n0 = blockIdx.x * 256 + (wave >> 1) * 64;

    float* attn_b = attn + (size_t)b * SEQ * SEQ;

    if (n0 > m0 + 31) {  // entire 32x64 region above diagonal -> attn == 0
#pragma unroll
        for (int mi = 0; mi < 2; ++mi)
#pragma unroll
            for (int nj = 0; nj < 4; ++nj)
#pragma unroll
                for (int r = 0; r < 8; ++r)
                    attn_b[(size_t)(m0 + mi * 16 + half * 8 + r) * SEQ +
                           n0 + nj * 16 + lo] = 0.0f;
        return;
    }

    const __bf16* q = qws + (size_t)b * SEQ * EMBED;
    const __bf16* k = kws + (size_t)b * SEQ * EMBED;

    v8f c[2][4] = {};
    for (int k0 = 0; k0 < EMBED; k0 += 32) {
        v16bf bfr[4];
#pragma unroll
        for (int nj = 0; nj < 4; ++nj) {
            // B[kk][n] = k[n0+nj*16+lo][k0+half*8+kk] (kk<8), [k0+16+half*8+kk-8] (kk>=8)
            const __bf16* brow =
                k + (size_t)(n0 + nj * 16 + lo) * EMBED + k0 + half * 8;
#pragma unroll
            for (int i = 0; i < 8; ++i) { bfr[nj][i] = brow[i]; bfr[nj][i + 8] = brow[16 + i]; }
        }
#pragma unroll
        for (int mi = 0; mi < 2; ++mi) {
            v16bf a;
            const __bf16* arow = q + (size_t)(m0 + mi * 16 + lo) * EMBED + k0;
#pragma unroll
            for (int i = 0; i < 8; ++i) {
                a[i]     = arow[half * 8 + i];
                a[i + 8] = arow[16 + half * 8 + i];
            }
#pragma unroll
            for (int nj = 0; nj < 4; ++nj)
                c[mi][nj] = WMMA_BF16(a, bfr[nj], c[mi][nj]);
        }
    }

#pragma unroll
    for (int mi = 0; mi < 2; ++mi)
#pragma unroll
        for (int nj = 0; nj < 4; ++nj)
#pragma unroll
            for (int r = 0; r < 8; ++r) {
                const int row = m0 + mi * 16 + half * 8 + r;
                const int col = n0 + nj * 16 + lo;
                float sv = c[mi][nj][r];
                if (col > row) sv = 0.0f;  // masked boundary entries
                attn_b[(size_t)row * SEQ + col] = sv;
            }
}

// ---------------------------------------------------------------------------
// Kernel 3: in-place causal softmax over each row's prefix [0, row], plus a
// bf16 copy of the normalized row (zero-padded to the next 64 boundary) for
// the attn@V GEMM.
// grid (SEQ, BATCH), block 256 (8 waves)
// ---------------------------------------------------------------------------
__global__ __launch_bounds__(256)
void softmax_kernel(float* __restrict__ attn, __bf16* __restrict__ attn_bf) {
    const int row = blockIdx.x;
    const int b   = blockIdx.y;
    float*  p  = attn    + ((size_t)b * SEQ + row) * SEQ;
    __bf16* pb = attn_bf + ((size_t)b * SEQ + row) * SEQ;
    const int L   = row + 1;                    // causal prefix length
    const int L2  = ((row >> 6) + 1) << 6;      // K4 reads up to 64-boundary
    const int tid = threadIdx.x;

    __shared__ float redmax[8];
    __shared__ float redsum[8];

    // pass 1: max
    float mx = -3.4e38f;
    for (int j = tid; j < L; j += 256) mx = fmaxf(mx, p[j]);
#pragma unroll
    for (int off = 16; off > 0; off >>= 1) mx = fmaxf(mx, __shfl_xor(mx, off, 32));
    if ((tid & 31) == 0) redmax[tid >> 5] = mx;
    __syncthreads();
    float m = redmax[0];
#pragma unroll
    for (int i = 1; i < 8; ++i) m = fmaxf(m, redmax[i]);

    // pass 2: exp + sum (store exp in place)
    float s = 0.0f;
    for (int j = tid; j < L; j += 256) {
        const float e = __expf(p[j] - m);
        p[j] = e;
        s += e;
    }
#pragma unroll
    for (int off = 16; off > 0; off >>= 1) s += __shfl_xor(s, off, 32);
    if ((tid & 31) == 0) redsum[tid >> 5] = s;
    __syncthreads();
    float tot = 0.0f;
#pragma unroll
    for (int i = 0; i < 8; ++i) tot += redsum[i];
    const float inv = 1.0f / tot;

    // pass 3: normalize (fp32 in place) + bf16 copy for the attn@V GEMM
    for (int j = tid; j < L; j += 256) {
        const float a = p[j] * inv;
        p[j]  = a;
        pb[j] = (__bf16)a;
    }
    // zero-pad bf16 row to the 64-aligned boundary read by out_kernel
    for (int j = L + tid; j < L2; j += 256) pb[j] = (__bf16)0.0f;
}

// ---------------------------------------------------------------------------
// Kernel 4: out = attn(bf16) @ v  (all fragments contiguous bf16 b128 loads).
// K loop stops at causal limit (bf16 attn zero-padded to 64 boundary).
// Wave tile: 64(M) x 32(N) -> 4 A + 2 B frags feed 8 WMMAs per step.
// grid (EMBED/64=8, SEQ/256=8, BATCH), block 256 (waves: 4 Msub x 2 Nsub)
// ---------------------------------------------------------------------------
__global__ __launch_bounds__(256)
void out_kernel(const __bf16* __restrict__ attn_bf,
                const __bf16* __restrict__ vtws,
                float* __restrict__ out) {
    const int lane = threadIdx.x & 31;
    const int wave = threadIdx.x >> 5;
    const int half = lane >> 4;
    const int lo   = lane & 15;

    const int b  = blockIdx.z;
    const int m0 = blockIdx.y * 256 + (wave & 3) * 64;
    const int n0 = blockIdx.x * 64  + (wave >> 2) * 32;

    const __bf16* A  = attn_bf + (size_t)b * SEQ * SEQ;
    const __bf16* VT = vtws    + (size_t)b * EMBED * SEQ;   // vT[n][s]

    const int kmax = m0 + 64;  // attn row m is zero-padded beyond col m

    v8f c[4][2] = {};
    for (int k0 = 0; k0 < kmax; k0 += 32) {
        v16bf bfr[2];
#pragma unroll
        for (int nj = 0; nj < 2; ++nj) {
            // B[kk][n] = vT[n0+nj*16+lo][k0+half*8+kk] (kk<8), [k0+16+half*8+kk-8] (kk>=8)
            const __bf16* brow =
                VT + (size_t)(n0 + nj * 16 + lo) * SEQ + k0 + half * 8;
#pragma unroll
            for (int i = 0; i < 8; ++i) { bfr[nj][i] = brow[i]; bfr[nj][i + 8] = brow[16 + i]; }
        }
#pragma unroll
        for (int mi = 0; mi < 4; ++mi) {
            v16bf a;
            const __bf16* arow = A + (size_t)(m0 + mi * 16 + lo) * SEQ + k0;
#pragma unroll
            for (int i = 0; i < 8; ++i) {
                a[i]     = arow[half * 8 + i];
                a[i + 8] = arow[16 + half * 8 + i];
            }
#pragma unroll
            for (int nj = 0; nj < 2; ++nj)
                c[mi][nj] = WMMA_BF16(a, bfr[nj], c[mi][nj]);
        }
    }

#pragma unroll
    for (int mi = 0; mi < 4; ++mi)
#pragma unroll
        for (int nj = 0; nj < 2; ++nj)
#pragma unroll
            for (int r = 0; r < 8; ++r)
                out[((size_t)b * SEQ + m0 + mi * 16 + half * 8 + r) * EMBED +
                    n0 + nj * 16 + lo] = c[mi][nj][r];
}

// ---------------------------------------------------------------------------
extern "C" void kernel_launch(void* const* d_in, const int* in_sizes, int n_in,
                              void* d_out, int out_size, void* d_ws, size_t ws_size,
                              hipStream_t stream) {
    (void)in_sizes; (void)n_in; (void)out_size; (void)ws_size;

    const float* x  = (const float*)d_in[0];
    const float* wq = (const float*)d_in[1];
    const float* bq = (const float*)d_in[2];
    const float* wk = (const float*)d_in[3];
    const float* bk = (const float*)d_in[4];
    const float* wv = (const float*)d_in[5];
    const float* bv = (const float*)d_in[6];

    float* out  = (float*)d_out;                         // [B,S,D]
    float* attn = out + (size_t)BATCH * SEQ * EMBED;     // [B,S,S]

    __bf16* ws_q  = (__bf16*)d_ws;                       // bf16 q (pre-scaled)
    __bf16* ws_k  = ws_q  + QK_ELEMS;                    // bf16 k
    __bf16* ws_vt = ws_k  + QK_ELEMS;                    // bf16 vT[b][n][s]
    __bf16* ws_wt = ws_vt + QK_ELEMS;                    // bf16 wT[3][n][k]
    __bf16* ws_ab = ws_wt + 3 * W_ELEMS;                 // bf16 attn copy

    wt_kernel<<<dim3((EMBED * EMBED) / 256, 1, 3), 256, 0, stream>>>(
        wq, wk, wv, ws_wt);

    qkv_proj_kernel<<<dim3(EMBED / 128, (BATCH * SEQ) / 128, 3), 256, 0, stream>>>(
        x, ws_wt, bq, bk, bv, ws_q);

    scores_kernel<<<dim3(SEQ / 256, SEQ / 64, BATCH), 256, 0, stream>>>(
        ws_q, ws_k, attn);

    softmax_kernel<<<dim3(SEQ, BATCH), 256, 0, stream>>>(attn, ws_ab);

    out_kernel<<<dim3(EMBED / 64, SEQ / 256, BATCH), 256, 0, stream>>>(
        ws_ab, ws_vt, out);
}
